// GroupedQueryAttention_6339371729728
// MI455X (gfx1250) — matrix operand order repaired and checked by reading, not compiled, not run on hardware
//
#include <hip/hip_runtime.h>
#include <hip/hip_bf16.h>

typedef __attribute__((ext_vector_type(16))) _Float16 v16h;
typedef __attribute__((ext_vector_type(8)))  _Float16 v8h;
typedef __attribute__((ext_vector_type(8)))  float    v8f;
typedef __attribute__((ext_vector_type(4)))  int      v4i;

static __device__ __forceinline__ v16h cat8(v8h a, v8h b) {
  return __builtin_shufflevector(a, b, 0,1,2,3,4,5,6,7,8,9,10,11,12,13,14,15);
}
static __device__ __forceinline__ v8f wmma_f16(v16h a, v16h b, v8f c) {
  return __builtin_amdgcn_wmma_f32_16x16x32_f16(false, a, false, b, (short)0, c,
                                                false, false);
}

#define MB   2
#define TT   2048
#define CC   1024
#define HH   16
#define DD   64
#define MM   (MB * TT)   // 4096

// ---- CDNA5 async global->LDS copy (ASYNCcnt path), with sync fallback ----
#if defined(__has_builtin)
#if __has_builtin(__builtin_amdgcn_global_load_async_to_lds_b128)
#define HAS_ASYNC 1
#endif
#endif
#ifndef HAS_ASYNC
#define HAS_ASYNC 0
#endif

#if HAS_ASYNC
typedef __attribute__((address_space(1))) v4i g_v4i;   // global int4
typedef __attribute__((address_space(3))) v4i l_v4i;   // LDS int4
#define ASYNC_CP16(dst, src)                                                   \
  __builtin_amdgcn_global_load_async_to_lds_b128((g_v4i*)(src), (l_v4i*)(dst), \
                                                 0, 0)
#else
#define ASYNC_CP16(dst, src) (*(v8h*)(dst) = *(const v8h*)(src))
#endif

// -------------------------------------------------------------------------
// fp32 -> f16 elementwise convert (one pass; x + all weights), 8 elems/thread
// -------------------------------------------------------------------------
__global__ __launch_bounds__(256)
void cvt_f16(const float* __restrict__ src, _Float16* __restrict__ dst, int n8) {
  const int i = blockIdx.x * 256 + threadIdx.x;
  if (i < n8) {
    const float* s = src + (size_t)i * 8;
    v8h h;
#pragma unroll
    for (int j = 0; j < 8; ++j) h[j] = (_Float16)s[j];
    *(v8h*)&dst[(size_t)i * 8] = h;
  }
}

// -------------------------------------------------------------------------
// Unified f16 GEMM: Y[m][n] = sum_k A[m][k] * W[n][k]
//   out_mode 0: f16 row-major Yh[m][n]                (Q, K projections)
//   out_mode 1: f16 V-transposed vt[(b*16+h)*64+d][T] (V projection)
//   out_mode 2: f32 Yf[m][n] + bias[n]                (output projection)
// 128 threads / 4 waves; block tile 128(M) x 64(N); wave tile 32(M) x 64(N)
// -> 8 WMMAs per K-step per wave fed by 12 ds_load_b128 (1.5:1), with two
// independent accumulator rows to cover XDL latency.
// Double-buffered LDS; tiles staged with async global->LDS b128 copies
// (6 async instructions per wave per K-step -> s_wait_asynccnt 6 keeps the
// next tile in flight while the current one is consumed).
// LDS rows padded to 72 halves (144B): conflict-free b128 fragment reads.
// -------------------------------------------------------------------------
__global__ __launch_bounds__(128)
void gemm_f16(const _Float16* __restrict__ A, const _Float16* __restrict__ Wt,
              _Float16* __restrict__ Yh, float* __restrict__ Yf,
              const float* __restrict__ bias, int Ksz, int Nsz, int out_mode) {
  __shared__ _Float16 Xl[2][128 * 72];
  __shared__ _Float16 Wl[2][64 * 72];
  const int tid  = threadIdx.x;
  const int lane = tid & 31;
  const int wv   = tid >> 5;           // 0..3, owns M rows wv*32 .. wv*32+31
  const int m0 = blockIdx.y * 128;
  const int n0 = blockIdx.x * 64;
  const int lo  = (lane < 16) ? 0 : 8;
  const int kbb = (lane < 16) ? 0 : 16;

  // 16-byte chunk assignment:
  // X tile 128x32 halves = 512 chunks (4/thread), W tile 64x32 = 256 (2/thread)
  const int xseg = (tid & 3) * 8;

  auto stage = [&](int buf, int k0) {
#pragma unroll
    for (int i = 0; i < 4; ++i) {
      const int xrow = (tid >> 2) + i * 32;
      ASYNC_CP16(&Xl[buf][xrow * 72 + xseg],
                 A + (size_t)(m0 + xrow) * Ksz + k0 + xseg);
    }
#pragma unroll
    for (int i = 0; i < 2; ++i) {
      const int wrow = (tid >> 2) + i * 32;
      ASYNC_CP16(&Wl[buf][wrow * 72 + xseg],
                 Wt + (size_t)(n0 + wrow) * Ksz + k0 + xseg);
    }
  };

  v8f acc[2][4] = {};
  const int S = Ksz / 32;
  stage(0, 0);
  for (int s = 0; s < S; ++s) {
    const int cur = s & 1;
    if (s + 1 < S) {
      stage(cur ^ 1, (s + 1) * 32);        // prefetch next tile (other buffer)
#if HAS_ASYNC
      asm volatile("s_wait_asynccnt 6" ::: "memory");   // current tile landed
#endif
    } else {
#if HAS_ASYNC
      asm volatile("s_wait_asynccnt 0" ::: "memory");
#endif
    }
    __syncthreads();

    v16h af[2];
#pragma unroll
    for (int mr = 0; mr < 2; ++mr) {
      const int am = wv * 32 + mr * 16 + (lane & 15);
      af[mr] = cat8(*(const v8h*)&Xl[cur][am * 72 + lo],
                    *(const v8h*)&Xl[cur][am * 72 + 16 + lo]);
    }
    v16h bf[4];
#pragma unroll
    for (int nn = 0; nn < 4; ++nn) {
      const int bn = nn * 16 + (lane & 15);
      bf[nn] = cat8(*(const v8h*)&Wl[cur][bn * 72 + lo],
                    *(const v8h*)&Wl[cur][bn * 72 + 16 + lo]);
    }
#pragma unroll
    for (int nn = 0; nn < 4; ++nn) {
      acc[0][nn] = wmma_f16(af[0], bf[nn], acc[0][nn]);
      acc[1][nn] = wmma_f16(af[1], bf[nn], acc[1][nn]);
    }
    __syncthreads();                        // done reading before re-staging
  }

#pragma unroll
  for (int mr = 0; mr < 2; ++mr) {
    const int mrow0 = m0 + wv * 32 + mr * 16 + ((lane < 16) ? 0 : 8);
    if (out_mode == 0) {
#pragma unroll
      for (int nn = 0; nn < 4; ++nn) {
        const int col = n0 + nn * 16 + (lane & 15);
#pragma unroll
        for (int r = 0; r < 8; ++r)
          Yh[(size_t)(mrow0 + r) * Nsz + col] = (_Float16)acc[mr][nn][r];
      }
    } else if (out_mode == 1) {
      const int bb = mrow0 >> 11;      // batch
      const int t  = mrow0 & 2047;     // time (8 consecutive rows stay in-batch)
#pragma unroll
      for (int nn = 0; nn < 4; ++nn) {
        const int col = n0 + nn * 16 + (lane & 15);
        const int hh = col >> 6, dd = col & 63;
        v8h p;
#pragma unroll
        for (int r = 0; r < 8; ++r) p[r] = (_Float16)acc[mr][nn][r];
        *(v8h*)&Yh[(size_t)((bb * 16 + hh) * 64 + dd) * TT + t] = p;  // b128
      }
    } else {
#pragma unroll
      for (int nn = 0; nn < 4; ++nn) {
        const int col = n0 + nn * 16 + (lane & 15);
        const float bv = bias[col];
#pragma unroll
        for (int r = 0; r < 8; ++r)
          Yf[(size_t)(mrow0 + r) * Nsz + col] = acc[mr][nn][r] + bv;
      }
    }
  }
}

// -------------------------------------------------------------------------
// Flash attention. 1 wave = one (b, h, 16-query tile). Keys in steps of 32.
// S = Q K^T in C/D layout (lane = key col, VGPR r = query row r + 8*(lane>=16))
// -> softmax row stats reduce with shfl_xor {1,2,4,8} inside 16-lane halves.
// P goes through a padded LDS tile (row stride 40 halves = 80B, 16B aligned,
// distinct banks) to be re-read as an A-fragment for P*V.
// -------------------------------------------------------------------------
__global__ __launch_bounds__(32)
void attn_kernel(const _Float16* __restrict__ Q, const _Float16* __restrict__ K,
                 const _Float16* __restrict__ Vt, _Float16* __restrict__ O) {
  __shared__ _Float16 P[16 * 40];
  const int lane = threadIdx.x;
  const int qt = blockIdx.x, hh = blockIdx.y, bb = blockIdx.z;
  const int q0 = qt * 16;
  const int nn = lane & 15;
  const int lo = (lane < 16) ? 0 : 8;
  const int kb = (lane < 16) ? 0 : 16;
  const int hi = (lane < 16) ? 0 : 8;
  const float NEG_INF = -__builtin_inff();

  // Q A-fragments (d = 0..31, 32..63), contiguous b128 chunks per lane
  const _Float16* qrow = Q + ((size_t)(bb * TT) + q0 + nn) * CC + hh * DD;
  v16h qf[2];
#pragma unroll
  for (int ds = 0; ds < 2; ++ds)
    qf[ds] = cat8(*(const v8h*)&qrow[ds * 32 + lo],
                  *(const v8h*)&qrow[ds * 32 + 16 + lo]);

  v8f o[4] = {};
  float mi[8], li[8];
#pragma unroll
  for (int r = 0; r < 8; ++r) { mi[r] = NEG_INF; li[r] = 0.f; }

  for (int kt = 0; kt <= q0 + 15; kt += 32) {
    if (kt + 32 <= q0 + 15) {  // prefetch next K/V tiles (global_prefetch_b8)
      __builtin_prefetch(K + ((size_t)(bb * TT) + kt + 32 + nn) * CC + hh * DD, 0, 0);
      __builtin_prefetch(Vt + ((size_t)((bb * 16 + hh) * 64 + nn)) * TT + kt + 32, 0, 0);
    }
    // ---- S = Q K^T for 32 keys (two 16-key N tiles) ----
    v8f s[2];
#pragma unroll
    for (int nb = 0; nb < 2; ++nb) {
      const _Float16* krow =
          K + ((size_t)(bb * TT) + kt + nb * 16 + nn) * CC + hh * DD;
      v16h k0f = cat8(*(const v8h*)&krow[lo],      *(const v8h*)&krow[16 + lo]);
      v16h k1f = cat8(*(const v8h*)&krow[32 + lo], *(const v8h*)&krow[48 + lo]);
      v8f z = {};
      z = wmma_f16(qf[0], k0f, z);
      z = wmma_f16(qf[1], k1f, z);
      s[nb] = z;
    }
    // ---- scale + causal mask ----
#pragma unroll
    for (int nb = 0; nb < 2; ++nb) {
      const int kg = kt + nb * 16 + nn;
#pragma unroll
      for (int r = 0; r < 8; ++r) {
        const int mg = q0 + r + hi;
        s[nb][r] = (kg <= mg) ? s[nb][r] * 0.125f : NEG_INF;
      }
    }
    // ---- online softmax update ----
    float alpha[8];
#pragma unroll
    for (int r = 0; r < 8; ++r) {
      float rm = fmaxf(s[0][r], s[1][r]);
      rm = fmaxf(rm, __shfl_xor(rm, 1));
      rm = fmaxf(rm, __shfl_xor(rm, 2));
      rm = fmaxf(rm, __shfl_xor(rm, 4));
      rm = fmaxf(rm, __shfl_xor(rm, 8));
      const float mn = fmaxf(mi[r], rm);      // finite from first block on
      alpha[r] = __expf(mi[r] - mn);
      mi[r] = mn;
      const float p0 = __expf(s[0][r] - mn);
      const float p1 = __expf(s[1][r] - mn);
      float rs = p0 + p1;
      rs += __shfl_xor(rs, 1);
      rs += __shfl_xor(rs, 2);
      rs += __shfl_xor(rs, 4);
      rs += __shfl_xor(rs, 8);
      li[r] = li[r] * alpha[r] + rs;
      P[(r + hi) * 40 + nn]      = (_Float16)p0;   // klocal = nn
      P[(r + hi) * 40 + 16 + nn] = (_Float16)p1;   // klocal = 16+nn
    }
#pragma unroll
    for (int dt = 0; dt < 4; ++dt)
#pragma unroll
      for (int r = 0; r < 8; ++r) o[dt][r] *= alpha[r];

    // single-wave LDS round trip: DS ops are in-order; ensure counter drained
    asm volatile("s_wait_dscnt 0" ::: "memory");

    // ---- P as A-fragment, V^T rows as B-fragments; O += P V ----
    v16h pf = cat8(*(const v8h*)&P[(lane & 15) * 40 + lo],
                   *(const v8h*)&P[(lane & 15) * 40 + 16 + lo]);
#pragma unroll
    for (int dt = 0; dt < 4; ++dt) {
      const _Float16* vrow =
          Vt + ((size_t)((bb * 16 + hh) * 64 + dt * 16 + nn)) * TT + kt;
      v16h vf = cat8(*(const v8h*)&vrow[lo], *(const v8h*)&vrow[16 + lo]);
      o[dt] = wmma_f16(pf, vf, o[dt]);
    }
  }

  // ---- normalize + store f16 row-major [b*T+t][C] for the Wo GEMM ----
#pragma unroll
  for (int r = 0; r < 8; ++r) {
    const float inv = 1.0f / li[r];
    const size_t row = (size_t)(bb * TT) + q0 + r + hi;
#pragma unroll
    for (int dt = 0; dt < 4; ++dt)
      O[row * CC + hh * DD + dt * 16 + nn] = (_Float16)(o[dt][r] * inv);
  }
}

// -------------------------------------------------------------------------
extern "C" void kernel_launch(void* const* d_in, const int* in_sizes, int n_in,
                              void* d_out, int out_size, void* d_ws, size_t ws_size,
                              hipStream_t stream) {
  const float* x  = (const float*)d_in[0];
  const float* Wq = (const float*)d_in[1];
  const float* Wk = (const float*)d_in[2];
  const float* Wv = (const float*)d_in[3];
  const float* Wo = (const float*)d_in[4];
  const float* bo = (const float*)d_in[5];

  const size_t xe = (size_t)MM * CC;   // 4M
  const size_t we = (size_t)CC * CC;   // 1M
  _Float16* xh  = (_Float16*)d_ws;     // [M][C] f16
  _Float16* wqh = xh + xe;
  _Float16* wkh = wqh + we;
  _Float16* wvh = wkh + we;
  _Float16* woh = wvh + we;
  _Float16* qb  = woh + we;            // [M][C]
  _Float16* kb  = qb + xe;             // [M][C]
  _Float16* vt  = kb + xe;             // [(b*16+h)*64+d][T]
  _Float16* at  = vt + xe;             // attention out, f16 row-major
  // total: 24M halves = 48MB of d_ws

  cvt_f16<<<dim3((int)(xe / 8 / 256)), 256, 0, stream>>>(x, xh, (int)(xe / 8));
  cvt_f16<<<dim3((int)(we / 8 / 256)), 256, 0, stream>>>(Wq, wqh, (int)(we / 8));
  cvt_f16<<<dim3((int)(we / 8 / 256)), 256, 0, stream>>>(Wk, wkh, (int)(we / 8));
  cvt_f16<<<dim3((int)(we / 8 / 256)), 256, 0, stream>>>(Wv, wvh, (int)(we / 8));
  cvt_f16<<<dim3((int)(we / 8 / 256)), 256, 0, stream>>>(Wo, woh, (int)(we / 8));

  dim3 gg(CC / 64, MM / 128);          // 16 x 32 blocks, 128 threads each
  gemm_f16<<<gg, 128, 0, stream>>>(xh, wqh, qb, nullptr, nullptr, CC, CC, 0);
  gemm_f16<<<gg, 128, 0, stream>>>(xh, wkh, kb, nullptr, nullptr, CC, CC, 0);
  gemm_f16<<<gg, 128, 0, stream>>>(xh, wvh, vt, nullptr, nullptr, CC, CC, 1);

  attn_kernel<<<dim3(TT / 16, HH, MB), 32, 0, stream>>>(qb, kb, vt, at);

  gemm_f16<<<gg, 128, 0, stream>>>(at, woh, nullptr, (float*)d_out, bo, CC, CC, 2);
}
